// MHA_6622839571242
// MI455X (gfx1250) — matrix operand order repaired and checked by reading, not compiled, not run on hardware
//
#include <hip/hip_runtime.h>
#include <hip/hip_bf16.h>

// ---------------------------------------------------------------------------
// MHA for MI455X / gfx1250 (wave32, WMMA 16x16x32 bf16, f32 accum)
//   pass 1: q/k/v = X @ W^T   (fp32 in -> bf16 [B,H,S,DK] out)
//   pass 2: flash-attention   (bf16 in -> bf16 [B,S,D] out)  [TDM K staging]
//   pass 3: out  = A @ Wo^T   (bf16 in -> fp32 [B,S,D] out)  [async LDS A staging]
// ---------------------------------------------------------------------------

typedef __bf16 bf16_t;
typedef bf16_t v16bf __attribute__((ext_vector_type(16)));
typedef float  v8f   __attribute__((ext_vector_type(8)));
typedef unsigned int u32x4 __attribute__((ext_vector_type(4)));
typedef int          i32x8 __attribute__((ext_vector_type(8)));
typedef int          i32x4 __attribute__((ext_vector_type(4)));

union Frag {
    v16bf        v;
    unsigned int u[8];
};

// ---- feature probes (device pass only) ------------------------------------
#if defined(__HIP_DEVICE_COMPILE__) && __has_builtin(__builtin_amdgcn_tensor_load_to_lds)
#define HAVE_TDM 1
#else
#define HAVE_TDM 0
#endif

#if defined(__HIP_DEVICE_COMPILE__) && __has_builtin(__builtin_amdgcn_global_load_async_to_lds_b128)
#define HAVE_GLB_ASYNC 1
#else
#define HAVE_GLB_ASYNC 0
#endif

#if HAVE_GLB_ASYNC
// Builtin expects: (v4i addrspace(1)* gsrc, v4i addrspace(3)* ldst, imm off, imm cpol)
typedef int v4i __attribute__((vector_size(16)));
typedef __attribute__((address_space(1))) v4i gv4i;
typedef __attribute__((address_space(3))) v4i sv4i;

__device__ __forceinline__ void async_b128(const void* gsrc, void* ldst) {
    // Generic ptr value == global address; low 32 bits of flat LDS ptr == LDS offset.
    __builtin_amdgcn_global_load_async_to_lds_b128(
        (gv4i*)(unsigned long long)(size_t)gsrc,
        (sv4i*)(unsigned int)(size_t)ldst,
        0, 0);
}
#endif

__device__ __forceinline__ unsigned short f2bf(float x) {
    union { float f; unsigned int u; } c; c.f = x;
    unsigned int u = c.u;
    unsigned int r = u + 0x7FFFu + ((u >> 16) & 1u);   // round-to-nearest-even
    return (unsigned short)(r >> 16);
}

__device__ __forceinline__ unsigned int pack2bf(float lo, float hi) {
#if defined(__HIP_DEVICE_COMPILE__) && __has_builtin(__builtin_amdgcn_cvt_pk_bf16_f32)
    typedef bf16_t v2bf __attribute__((ext_vector_type(2)));
    union { v2bf v; unsigned int u; } r;
    r.v = __builtin_amdgcn_cvt_pk_bf16_f32(lo, hi);
    return r.u;
#else
    return (unsigned int)f2bf(lo) | ((unsigned int)f2bf(hi) << 16);
#endif
}

__device__ __forceinline__ void wait_tensor_zero() {
#if defined(__HIP_DEVICE_COMPILE__)
#if __has_builtin(__builtin_amdgcn_s_wait_tensorcnt)
    __builtin_amdgcn_s_wait_tensorcnt(0);
#else
    asm volatile("s_wait_tensorcnt 0x0" ::: "memory");
#endif
#endif
}

__device__ __forceinline__ void wait_async_zero() {
#if defined(__HIP_DEVICE_COMPILE__)
#if __has_builtin(__builtin_amdgcn_s_wait_asynccnt)
    __builtin_amdgcn_s_wait_asynccnt(0);
#else
    asm volatile("s_wait_asynccnt 0x0" ::: "memory");
#endif
#endif
}

constexpr int Bc  = 2;
constexpr int Sc  = 2048;
constexpr int Dc  = 768;
constexpr int Hc  = 12;
constexpr int DKc = 64;
constexpr int Mrows = Bc * Sc;            // 4096

// ---------------------------------------------------------------------------
// GEMM: C[M x 768] = A[M x 768] * W^T   (W is [768 out][768 in], row-major)
// Block tile 128x64, K-step 32.  8 waves, each wave -> 32x32 (2x2 WMMA tiles).
// OUT_HEADSPLIT: write bf16 into [B,H,S,DK]; else fp32 row-major [M,768].
// ---------------------------------------------------------------------------
template <bool A_BF16, bool OUT_HEADSPLIT>
__global__ void __launch_bounds__(256)
gemm_proj(const void* __restrict__ Ain,
          const float* __restrict__ W,
          void* __restrict__ Cout)
{
    constexpr int BM = 128, BN = 64, KT = 32, PITCH = 40; // halves; 80B rows (16B aligned)
    __shared__ unsigned short As[BM * PITCH];
    __shared__ unsigned short Bs[BN * PITCH];

    const int tid   = threadIdx.x;
    const int lane  = tid & 31;
    const int wave  = tid >> 5;
    const int half  = lane >> 4;
    const int l16   = lane & 15;
    const int waveM = wave & 3;     // 4 wave rows of 32
    const int waveN = wave >> 2;    // 2 wave cols of 32
    const int bm    = blockIdx.x * BM;
    const int bn    = blockIdx.y * BN;

    const float*          Af = (const float*)Ain;
    const unsigned short* Ab = (const unsigned short*)Ain;

    v8f acc[2][2];
    {
        v8f z = {};
        acc[0][0] = z; acc[0][1] = z; acc[1][0] = z; acc[1][1] = z;
    }

    for (int k0 = 0; k0 < Dc; k0 += KT) {
        // ---- stage A tile (128 x 32) as bf16 ----
        {
            const int r = tid >> 1;            // 0..127
            const int c = (tid & 1) * 16;      // 0 / 16
            unsigned short* dst = &As[r * PITCH + c];
            if constexpr (A_BF16) {
                const unsigned short* src = Ab + (size_t)(bm + r) * Dc + k0 + c;
#if HAVE_GLB_ASYNC
                async_b128(src, dst);
                async_b128(src + 8, dst + 8);
#else
                const unsigned int* s32 = (const unsigned int*)src;
                unsigned int* d32 = (unsigned int*)dst;
#pragma unroll
                for (int i = 0; i < 8; ++i) d32[i] = s32[i];
#endif
            } else {
                const float* src = Af + (size_t)(bm + r) * Dc + k0 + c;
                unsigned int* d32 = (unsigned int*)dst;
#pragma unroll
                for (int i = 0; i < 8; ++i)
                    d32[i] = pack2bf(src[2 * i], src[2 * i + 1]);
            }
        }
        // ---- stage B tile: Bs[n][k] = W[bn+n][k0+k]  (this IS B^T layout) ----
        {
            const int n = tid >> 2;            // 0..63
            const int c = (tid & 3) * 8;       // 0,8,16,24
            const float* src = W + (size_t)(bn + n) * Dc + k0 + c;
            unsigned int* d32 = (unsigned int*)&Bs[n * PITCH + c];
#pragma unroll
            for (int i = 0; i < 4; ++i)
                d32[i] = pack2bf(src[2 * i], src[2 * i + 1]);
        }
#if HAVE_GLB_ASYNC
        if constexpr (A_BF16) wait_async_zero();
#endif
        __syncthreads();

        // ---- fragments ----
        Frag a[2], b[2];
#pragma unroll
        for (int mt = 0; mt < 2; ++mt) {
            const int m = waveM * 32 + mt * 16 + l16;
            const unsigned short* row = &As[m * PITCH];
#pragma unroll
            for (int j = 0; j < 4; ++j) {
                const int ka = half * 16 + 2 * j;         // contiguous K: k = 16*half + s
                const int kb = half * 16 + 8 + 2 * j;
                a[mt].u[j]     = *(const unsigned int*)(row + ka);
                a[mt].u[4 + j] = *(const unsigned int*)(row + kb);
            }
        }
#pragma unroll
        for (int nt = 0; nt < 2; ++nt) {
            const int n  = waveN * 32 + nt * 16 + l16;
            const int kh = half * 16;
            const unsigned short* row = &Bs[n * PITCH + kh];
#pragma unroll
            for (int j = 0; j < 8; ++j)
                b[nt].u[j] = *(const unsigned int*)(row + 2 * j);
        }
#pragma unroll
        for (int mt = 0; mt < 2; ++mt)
#pragma unroll
            for (int nt = 0; nt < 2; ++nt)
                acc[mt][nt] = __builtin_amdgcn_wmma_f32_16x16x32_bf16(
                    false, a[mt].v, false, b[nt].v, (short)0, acc[mt][nt],
                    false, false);
        __syncthreads();
    }

    // ---- epilogue ----
#pragma unroll
    for (int mt = 0; mt < 2; ++mt) {
#pragma unroll
        for (int nt = 0; nt < 2; ++nt) {
#pragma unroll
            for (int r = 0; r < 8; ++r) {
                const int m = bm + waveM * 32 + mt * 16 + r + 8 * half;
                const int n = bn + waveN * 32 + nt * 16 + l16;
                const float val = acc[mt][nt][r];
                if constexpr (OUT_HEADSPLIT) {
                    const int h  = n >> 6;
                    const int dk = n & 63;
                    const int bi = m >> 11;     // m / S
                    const int s  = m & 2047;    // m % S
                    ((unsigned short*)Cout)
                        [(((size_t)bi * Hc + h) * Sc + s) * DKc + dk] = f2bf(val);
                } else {
                    ((float*)Cout)[(size_t)m * Dc + n] = val;
                }
            }
        }
    }
}

// ---------------------------------------------------------------------------
// Flash attention.  Grid (S/128, H, B), 256 threads = 8 waves.
// Wave w owns q rows [q0+16w, q0+16w+16).  KV blocks of 64 rows.
// K tile staged by TDM (tensor_load_to_lds) with native LDS row padding.
// ---------------------------------------------------------------------------
__global__ void __launch_bounds__(256)
attn_kernel(const unsigned short* __restrict__ q,   // bf16 [B,H,S,64]
            const unsigned short* __restrict__ k,   // bf16 [B,H,S,64]
            const unsigned short* __restrict__ v,   // bf16 [B,H,S,64]
            const int* __restrict__ mask,           // i32  [B,1,S,S]
            unsigned short* __restrict__ outw)      // bf16 [B,S,768]
{
    constexpr int KB   = 64;
    constexpr int KPIT = 72;   // 144B rows: TDM pad 4 DWORDs per 32 DWORDs; conflict-free
    constexpr int VPIT = 66;
    constexpr int PPIT = 66;
    __shared__ unsigned short Ks[KB  * KPIT];       // [kv][d]
    __shared__ unsigned short Vt[DKc * VPIT];       // [d][kv]   (transposed V)
    __shared__ unsigned short Pb[8 * 16 * PPIT];    // per-wave P strips

    const int tid  = threadIdx.x;
    const int lane = tid & 31;
    const int wave = tid >> 5;
    const int half = lane >> 4;
    const int l16  = lane & 15;

    const int qt = blockIdx.x;
    const int h  = blockIdx.y;
    const int b  = blockIdx.z;
    const int q0 = qt * 128 + wave * 16;

    const size_t headOff = ((size_t)b * Hc + h) * Sc * DKc;
    const unsigned short* qh = q + headOff;
    const unsigned short* kh = k + headOff;
    const unsigned short* vh = v + headOff;
    const int*            mb = mask + (size_t)b * Sc * Sc;

    const float NEG_INF = -__builtin_inff();

    // ---- Q fragments (persist in registers) ----
    Frag aq[2];
    {
        const unsigned short* qr = qh + (size_t)(q0 + l16) * DKc;
#pragma unroll
        for (int ks = 0; ks < 2; ++ks) {
#pragma unroll
            for (int j = 0; j < 4; ++j) {
                const int ka = ks * 32 + half * 16 + 2 * j;
                const int kb = ks * 32 + half * 16 + 8 + 2 * j;
                aq[ks].u[j]     = *(const unsigned int*)(qr + ka);
                aq[ks].u[4 + j] = *(const unsigned int*)(qr + kb);
            }
        }
    }

    float mrow[8], lrow[8];
    v8f   o[4];
    {
        v8f z = {};
#pragma unroll
        for (int nt = 0; nt < 4; ++nt) o[nt] = z;
#pragma unroll
        for (int r = 0; r < 8; ++r) { mrow[r] = NEG_INF; lrow[r] = 0.f; }
    }

    for (int kb = 0; kb < Sc / KB; ++kb) {
        const int kv0 = kb * KB;

        // ---- stage K tile [kv][d] ----
#if HAVE_TDM
        if (wave == 0) {
            const unsigned long long ga =
                (unsigned long long)(size_t)(kh + (size_t)kv0 * DKc);
            const unsigned int ldsa = (unsigned int)(size_t)(const void*)&Ks[0];
            u32x4 g0;
            g0.x = 1u;                                   // count=1 (valid D#)
            g0.y = ldsa;                                 // lds_addr (bytes)
            g0.z = (unsigned int)ga;                     // global_addr[31:0]
            g0.w = (unsigned int)((ga >> 32) & 0x01FFFFFFu) | (2u << 30); // [56:32] | type=2
            i32x8 g1;
            g1[0] = (int)((1u << 16)      // data_size = 2B
                        | (1u << 20)      // pad_enable
                        | (4u << 22)      // pad_interval: 32 DWORDs
                        | (3u << 25));    // pad_amount:   4 DWORDs
            g1[1] = (int)(64u << 16);     // tensor_dim0[15:0] (=64 elems/row)
            g1[2] = (int)(64u << 16);     // tensor_dim0[31:16]=0 | tensor_dim1[15:0]=64
            g1[3] = (int)(64u << 16);     // tensor_dim1[31:16]=0 | tile_dim0=64
            g1[4] = 64;                   // tile_dim1=64 | tile_dim2=0
            g1[5] = 64;                   // tensor_dim0_stride[31:0] = 64 elems
            g1[6] = 0;                    // stride0 hi | stride1 lo
            g1[7] = 0;                    // stride1 hi
            i32x4 z4 = {0, 0, 0, 0};
            i32x8 z8 = {0, 0, 0, 0, 0, 0, 0, 0};
            __builtin_amdgcn_tensor_load_to_lds(g0, g1, z4, z4, z8, 0);
            wait_tensor_zero();
        }
#else
        {
            const int r = tid >> 2;            // 0..63
            const int c = (tid & 3) * 16;      // 0..48
            const unsigned int* s32 =
                (const unsigned int*)(kh + (size_t)(kv0 + r) * DKc + c);
            unsigned int* d32 = (unsigned int*)&Ks[r * KPIT + c];
#pragma unroll
            for (int i = 0; i < 8; ++i) d32[i] = s32[i];
        }
#endif
        // ---- stage V tile transposed: Vt[d][kv] ----
        {
            const int r = tid >> 2;            // kv local
            const int c = (tid & 3) * 16;      // d base
            const unsigned short* s = vh + (size_t)(kv0 + r) * DKc + c;
#pragma unroll
            for (int i = 0; i < 16; ++i) Vt[(c + i) * VPIT + r] = s[i];
        }
        __syncthreads();

        // ---- scores: S = Q(16x64) * K^T(64x64) ----
        v8f sc[4];
        {
            v8f z = {};
#pragma unroll
            for (int nt = 0; nt < 4; ++nt) sc[nt] = z;
        }
#pragma unroll
        for (int nt = 0; nt < 4; ++nt) {
            const int n = nt * 16 + l16;       // kv column
#pragma unroll
            for (int ks = 0; ks < 2; ++ks) {
                Frag bk;
                const unsigned short* row = &Ks[n * KPIT + ks * 32 + half * 16];
#pragma unroll
                for (int j = 0; j < 8; ++j)
                    bk.u[j] = *(const unsigned int*)(row + 2 * j);
                sc[nt] = __builtin_amdgcn_wmma_f32_16x16x32_bf16(
                    false, aq[ks].v, false, bk.v, (short)0, sc[nt], false, false);
            }
        }

        // ---- scale + mask + online softmax (16-lane row groups) ----
#pragma unroll
        for (int r = 0; r < 8; ++r) {
            const int row = q0 + r + 8 * half;
            const int* mrp = mb + (size_t)row * Sc + kv0;
            if (kv0 + KB < Sc)
                __builtin_prefetch(mrp + KB, 0, 0);   // next block's mask row
            float vals[4];
            float bmax = NEG_INF;
#pragma unroll
            for (int nt = 0; nt < 4; ++nt) {
                const int col = nt * 16 + l16;
                float x = sc[nt][r] * 0.125f;      // 1/sqrt(64)
                if (mrp[col] == 0) x = NEG_INF;
                vals[nt] = x;
                bmax = fmaxf(bmax, x);
            }
#pragma unroll
            for (int off = 8; off >= 1; off >>= 1)
                bmax = fmaxf(bmax, __shfl_xor(bmax, off, 32));

            const float nm    = fmaxf(mrow[r], bmax);
            const float alpha = (nm == NEG_INF) ? 1.0f : __expf(mrow[r] - nm);
            float psum = 0.f;
#pragma unroll
            for (int nt = 0; nt < 4; ++nt) {
                const float p = (nm == NEG_INF) ? 0.f : __expf(vals[nt] - nm);
                psum += p;
                Pb[(wave * 16 + r + 8 * half) * PPIT + nt * 16 + l16] = f2bf(p);
            }
#pragma unroll
            for (int off = 8; off >= 1; off >>= 1)
                psum += __shfl_xor(psum, off, 32);

            lrow[r] = lrow[r] * alpha + psum;
            mrow[r] = nm;
#pragma unroll
            for (int nt = 0; nt < 4; ++nt) o[nt][r] *= alpha;
        }

        // ---- O += P(16x64) * V(64x64)  (P re-read in A-frag layout) ----
#pragma unroll
        for (int ks2 = 0; ks2 < 2; ++ks2) {
            Frag ap;
            {
                const unsigned short* pr = &Pb[(wave * 16 + l16) * PPIT];
#pragma unroll
                for (int j = 0; j < 4; ++j) {
                    const int ka  = ks2 * 32 + half * 16 + 2 * j;
                    const int kb2 = ks2 * 32 + half * 16 + 8 + 2 * j;
                    ap.u[j]     = *(const unsigned int*)(pr + ka);
                    ap.u[4 + j] = *(const unsigned int*)(pr + kb2);
                }
            }
#pragma unroll
            for (int nt = 0; nt < 4; ++nt) {
                Frag bv;
                const int n = nt * 16 + l16;   // dk column
                const unsigned short* row = &Vt[n * VPIT + ks2 * 32 + half * 16];
#pragma unroll
                for (int j = 0; j < 8; ++j)
                    bv.u[j] = *(const unsigned int*)(row + 2 * j);
                o[nt] = __builtin_amdgcn_wmma_f32_16x16x32_bf16(
                    false, ap.v, false, bv.v, (short)0, o[nt], false, false);
            }
        }
        __syncthreads();
    }

    // ---- normalize + write [B,S,768] bf16 (heads re-interleaved) ----
#pragma unroll
    for (int r = 0; r < 8; ++r) {
        const float inv = (lrow[r] > 0.f) ? (1.0f / lrow[r]) : 0.f;
        const int row = q0 + r + 8 * half;
        unsigned short* orow = outw + ((size_t)b * Sc + row) * Dc + h * DKc;
#pragma unroll
        for (int nt = 0; nt < 4; ++nt)
            orow[nt * 16 + l16] = f2bf(o[nt][r] * inv);
    }
}

// ---------------------------------------------------------------------------
// Launch
// ---------------------------------------------------------------------------
extern "C" void kernel_launch(void* const* d_in, const int* in_sizes, int n_in,
                              void* d_out, int out_size, void* d_ws,
                              size_t ws_size, hipStream_t stream)
{
    (void)in_sizes; (void)n_in; (void)out_size; (void)ws_size;

    const float* Qin  = (const float*)d_in[0];
    const float* Kin  = (const float*)d_in[1];
    const float* Vin  = (const float*)d_in[2];
    const int*   mask = (const int*)d_in[3];
    const float* WQ   = (const float*)d_in[4];
    const float* WK   = (const float*)d_in[5];
    const float* WV   = (const float*)d_in[6];
    const float* WO   = (const float*)d_in[7];
    float*       out  = (float*)d_out;

    const size_t headElems = (size_t)Bc * Hc * Sc * DKc;   // 3,145,728
    unsigned short* qw = (unsigned short*)d_ws;
    unsigned short* kw = qw + headElems;
    unsigned short* vw = kw + headElems;
    unsigned short* aw = vw + headElems;                   // [B,S,768] bf16

    dim3 gg(Mrows / 128, Dc / 64);   // (32, 12)
    dim3 bb(256);

    gemm_proj<false, true><<<gg, bb, 0, stream>>>((const void*)Qin, WQ, (void*)qw);
    gemm_proj<false, true><<<gg, bb, 0, stream>>>((const void*)Kin, WK, (void*)kw);
    gemm_proj<false, true><<<gg, bb, 0, stream>>>((const void*)Vin, WV, (void*)vw);

    attn_kernel<<<dim3(Sc / 128, Hc, Bc), 256, 0, stream>>>(qw, kw, vw, mask, aw);

    gemm_proj<true, false><<<gg, bb, 0, stream>>>((const void*)aw, WO, (void*)out);
}
